// ModifiedSRCMLayer_5145370821188
// MI455X (gfx1250) — matrix operand order repaired and checked by reading, not compiled, not run on hardware
//
#include <hip/hip_runtime.h>
#include <hip/hip_bf16.h>

// ---------------------------------------------------------------------------
// Bidirectional vision-Mamba layer for MI455X (gfx1250, wave32).
//
// Dataflow (all f32 in workspace, GEMMs run through v_wmma_f32_16x16x32_f16
// with on-the-fly f32->f16 conversion, f32 accumulation):
//   1. prep:   pos-embed 2x bilinear downsample + add + LayerNorm  -> xs, xn
//   2. gate:   WMMA GEMM xn @ gate_w^T + b, sigmoid                -> gate
//   3. inproj: WMMA GEMM (batched over G=4) xn_g @ in_w^T          -> xz
//   4. conv:   causal depthwise conv(k=4) + SiLU (f and reversed b)-> xc
//   5. xproj:  WMMA GEMM (batched G, per dir) xc @ xpw^T (N=36)    -> p
//   6. dt:     p[:, :4] @ dtw^T + b, softplus                      -> dt
//   7. scan:   sequential selective scan, one lane per (g,b,d),
//              two launches (fwd writes, bwd accumulates)          -> y
//   8. outproj:WMMA GEMM (batched G) y @ ow^T, interleaved store   -> x_mamba
//   9. combine:gate*x_mamba + (1-gate)*xs                          -> x_out
//  10. proj:   WMMA GEMM x_out @ proj_w^T + b, NCHW-transposed     -> d_out
//
// GEMM inner loop: per wave one 16x32 output tile, A fragment reused across
// two WMMAs; all fragment loads are contiguous float4 (global_load_b128),
// padded-N handled by clamping the B row pointer (stores are guarded), so
// EXEC stays full and no divergent loads appear in the hot loop.
// ---------------------------------------------------------------------------

typedef __attribute__((ext_vector_type(16))) _Float16 v16h;
typedef __attribute__((ext_vector_type(8)))  float    v8f;

#define D_STATE 16
#define D_INNER 128
#define D_MODEL 64
#define NGROUP  4
#define LSEQ    1024
#define NBATCH  4
#define NTOK    (NBATCH * LSEQ)       // 4096
#define CDIM    256

// -------------------------------------------------------------------- prep
__global__ void __launch_bounds__(256)
srcm_prep_ln(const float* __restrict__ x, const float* __restrict__ pos,
             const float* __restrict__ nw, const float* __restrict__ nb,
             float* __restrict__ xs, float* __restrict__ xn) {
  int token = blockIdx.x;            // 0..4095  (b*1024 + l)
  int b = token >> 10;
  int l = token & 1023;
  int h = l >> 5, w = l & 31;
  int c = threadIdx.x;               // 256 channels

  // 2x bilinear downsample of 64x64 pos grid: sample at 2h+0.5, 2w+0.5
  int r0 = (2 * h) * 64 + 2 * w;
  int r1 = (2 * h + 1) * 64 + 2 * w;
  float pe = 0.25f * (pos[(long)r0 * CDIM + c] + pos[(long)(r0 + 1) * CDIM + c] +
                      pos[(long)r1 * CDIM + c] + pos[(long)(r1 + 1) * CDIM + c]);
  float xv = x[((long)(b * CDIM + c) << 10) + l] + pe;
  xs[(long)token * CDIM + c] = xv;

  __shared__ float red[256];
  red[c] = xv;
  __syncthreads();
  for (int s = 128; s > 0; s >>= 1) { if (c < s) red[c] += red[c + s]; __syncthreads(); }
  float mu = red[0] * (1.0f / 256.0f);
  __syncthreads();
  float dv = xv - mu;
  red[c] = dv * dv;
  __syncthreads();
  for (int s = 128; s > 0; s >>= 1) { if (c < s) red[c] += red[c + s]; __syncthreads(); }
  float var = red[0] * (1.0f / 256.0f);
  float rs = rsqrtf(var + 1e-5f);
  xn[(long)token * CDIM + c] = dv * rs * nw[c] + nb[c];
}

// ---------------------------------------------------------- WMMA GEMM core
// C[m][n] = act( sum_k A[m][k] * W[n][k] + bias[n] )
// A: MxK row-major (lda), W: NxK row-major (ldw). One 16x32 tile per wave.
// act: 0 = none, 1 = sigmoid.  out_nchw: store to (B, N, 32, 32) layout.
__global__ void __launch_bounds__(256)
srcm_wmma_gemm(const float* __restrict__ A, const float* __restrict__ W,
               const float* __restrict__ bias, float* __restrict__ C,
               int M, int N, int K, int lda, int ldw, int ldc,
               long strideA, long strideW, long strideC,
               int act, int out_nchw) {
  int batch = blockIdx.y;
  const float* Ab = A + (long)batch * strideA;
  const float* Wb = W + (long)batch * strideW;
  float*       Cb = C + (long)batch * strideC;

  int wave   = threadIdx.x >> 5;
  int lane   = threadIdx.x & 31;
  int lane15 = lane & 15;
  int laneHi = (lane >> 4) & 1;
  int ntn    = (N + 31) >> 5;             // 32-wide N tiles
  int tiles  = (M >> 4) * ntn;
  int tile   = blockIdx.x * 8 + wave;
  if (tile >= tiles) return;              // wave-uniform exit (EXEC stays full)
  int tm = tile / ntn;
  int tn = tile - tm * ntn;

  int m  = (tm << 4) + lane15;            // M is always a multiple of 16 here
  int n0 = (tn << 5) + lane15;
  int n1 = n0 + 16;
  // Clamp padded B rows in-bounds: garbage only reaches unstored columns.
  int n0c = (n0 < N) ? n0 : (N - 1);
  int n1c = (n1 < N) ? n1 : (N - 1);
  const float4* Arow = (const float4*)(Ab + (long)m * lda);
  const float4* W0   = (const float4*)(Wb + (long)n0c * ldw);
  const float4* W1   = (const float4*)(Wb + (long)n1c * ldw);

  v8f acc0 = {};
  v8f acc1 = {};
#pragma unroll 2
  for (int k0 = 0; k0 < K; k0 += 32) {
    // A f16 16x32 layout: lane holds K = k0+laneHi*16 + {0..15}
    int ab = (k0 + (laneHi << 4)) >> 2;       // float4 index
    float4 a0 = Arow[ab];
    float4 a1 = Arow[ab + 1];
    float4 a2 = Arow[ab + 2];
    float4 a3 = Arow[ab + 3];
    // B f16 32x16 layout: lane holds K = k0+laneHi*16 + {0..15}
    int bb = (k0 + (laneHi << 4)) >> 2;
    float4 b0 = W0[bb], b1 = W0[bb + 1], b2 = W0[bb + 2], b3 = W0[bb + 3];
    float4 c0 = W1[bb], c1 = W1[bb + 1], c2 = W1[bb + 2], c3 = W1[bb + 3];

    float at[16]  = {a0.x, a0.y, a0.z, a0.w, a1.x, a1.y, a1.z, a1.w,
                     a2.x, a2.y, a2.z, a2.w, a3.x, a3.y, a3.z, a3.w};
    float bt0[16] = {b0.x, b0.y, b0.z, b0.w, b1.x, b1.y, b1.z, b1.w,
                     b2.x, b2.y, b2.z, b2.w, b3.x, b3.y, b3.z, b3.w};
    float bt1[16] = {c0.x, c0.y, c0.z, c0.w, c1.x, c1.y, c1.z, c1.w,
                     c2.x, c2.y, c2.z, c2.w, c3.x, c3.y, c3.z, c3.w};
    v16h af, bf0, bf1;
#pragma unroll
    for (int e = 0; e < 16; ++e) {
      af[e]  = (_Float16)at[e];
      bf0[e] = (_Float16)bt0[e];
      bf1[e] = (_Float16)bt1[e];
    }
    acc0 = __builtin_amdgcn_wmma_f32_16x16x32_f16(false, af, false, bf0,
                                                  (short)0, acc0, false, false);
    acc1 = __builtin_amdgcn_wmma_f32_16x16x32_f16(false, af, false, bf1,
                                                  (short)0, acc1, false, false);
  }

#pragma unroll
  for (int half = 0; half < 2; ++half) {
    v8f acc = half ? acc1 : acc0;
#pragma unroll
    for (int r = 0; r < 8; ++r) {
      int mm = (tm << 4) + r + (laneHi << 3);
      int nn = (tn << 5) + (half << 4) + lane15;
      if (nn < N) {
        float v = acc[r];
        if (bias) v += bias[nn];
        if (act == 1) v = 1.0f / (1.0f + expf(-v));
        if (out_nchw) {
          // tokens are (b*1024 + l); store out[b][nn][l]
          long idx = ((long)((mm >> 10) * N + nn) << 10) + (mm & 1023);
          Cb[idx] = v;
        } else {
          Cb[(long)mm * ldc + nn] = v;
        }
      }
    }
  }
}

// ----------------------------------------------------- causal conv + SiLU
// xz layout: [g][4096][256]  (x_pre = cols 0..127, z = cols 128..255)
// xc layout: [gd][4096][128], gd = g*2+dir; dir=1 processes reversed sequence.
__global__ void __launch_bounds__(256)
srcm_conv_silu(const float* __restrict__ xz,
               const float* __restrict__ cwf, const float* __restrict__ cbf,
               const float* __restrict__ cwb, const float* __restrict__ cbb,
               float* __restrict__ xc) {
  int idx = blockIdx.x * 256 + threadIdx.x;      // total = 8*4096*128 = 2^22
  int d  = idx & 127;
  int m  = (idx >> 7) & 4095;
  int gd = idx >> 19;
  int g = gd >> 1, dir = gd & 1;
  int b = m >> 10, l = m & 1023;

  const float* wv = (dir ? cwb : cwf) + (long)(g * D_INNER + d) * 4;
  float acc = (dir ? cbb : cbf)[g * D_INNER + d];
#pragma unroll
  for (int j = 0; j < 4; ++j) {
    int li = l - 3 + j;
    if (li >= 0) {
      int lsrc = dir ? (LSEQ - 1 - li) : li;
      acc += wv[j] * xz[((long)g * NTOK + b * LSEQ + lsrc) * CDIM + d];
    }
  }
  xc[idx] = acc / (1.0f + expf(-acc));           // SiLU
}

// --------------------------------------------------------- dt = softplus
// p layout: [gd][4096][36]; dt layout: [gd][4096][128]
__global__ void __launch_bounds__(256)
srcm_dt(const float* __restrict__ p,
        const float* __restrict__ dtwf, const float* __restrict__ dtbf,
        const float* __restrict__ dtwb, const float* __restrict__ dtbb,
        float* __restrict__ dt) {
  int idx = blockIdx.x * 256 + threadIdx.x;
  int d  = idx & 127;
  int m  = (idx >> 7) & 4095;
  int gd = idx >> 19;
  int g = gd >> 1, dir = gd & 1;
  const float* pr = p + ((long)gd * NTOK + m) * 36;
  const float* wv = (dir ? dtwb : dtwf) + (long)(g * D_INNER + d) * 4;
  float v = (dir ? dtbb : dtbf)[g * D_INNER + d] +
            pr[0] * wv[0] + pr[1] * wv[1] + pr[2] * wv[2] + pr[3] * wv[3];
  dt[idx] = (v > 20.0f) ? v : log1pf(expf(v));
}

// ------------------------------------------------------- selective scan
// One lane per (g,b,d): 4*4*128 = 2048 lanes, 16-state recurrence over L=1024.
// dir=0 writes y; dir=1 accumulates (separate launch -> stream-ordered).
__global__ void __launch_bounds__(256)
srcm_scan(const float* __restrict__ xc, const float* __restrict__ p,
          const float* __restrict__ dt, const float* __restrict__ xz,
          const float* __restrict__ A_log, const float* __restrict__ Dp,
          float* __restrict__ y, int dir) {
  int t = blockIdx.x * 256 + threadIdx.x;
  if (t >= 2048) return;
  int d = t & 127, b = (t >> 7) & 3, g = (t >> 9) & 3;
  int gd = g * 2 + dir;

  float Arow[D_STATE];
#pragma unroll
  for (int n = 0; n < D_STATE; ++n)
    Arow[n] = -__expf(A_log[(long)(g * D_INNER + d) * D_STATE + n]);
  float Dv = Dp[g * D_INNER + d];

  float h[D_STATE];
#pragma unroll
  for (int n = 0; n < D_STATE; ++n) h[n] = 0.0f;

  long base = (long)gd * NTOK;
  for (int l = 0; l < LSEQ; ++l) {
    long mi = base + b * LSEQ + l;
    float delta = dt[mi * D_INNER + d];
    float u     = xc[mi * D_INNER + d];
    const float* pr = p + mi * 36;
    float du = delta * u;
    float acc = 0.0f;
#pragma unroll
    for (int n = 0; n < D_STATE; ++n) {
      float dA = __expf(delta * Arow[n]);
      h[n] = dA * h[n] + du * pr[4 + n];
      acc += h[n] * pr[20 + n];
    }
    float yv = acc + u * Dv;
    int lo = dir ? (LSEQ - 1 - l) : l;
    long mo = (long)b * LSEQ + lo;
    float zv = xz[((long)g * NTOK + mo) * CDIM + D_INNER + d];
    float gz = zv / (1.0f + __expf(-zv));          // SiLU(z)
    long oi = ((long)g * NTOK + mo) * D_INNER + d;
    float val = yv * gz;
    if (dir) y[oi] += val; else y[oi] = val;
  }
}

// --------------------------------------------------------------- combine
__global__ void __launch_bounds__(256)
srcm_combine(const float* __restrict__ gate, const float* __restrict__ xm,
             const float* __restrict__ xs, float* __restrict__ xo) {
  int idx = blockIdx.x * 256 + threadIdx.x;      // 4096*256
  float gv = gate[idx];
  xo[idx] = gv * xm[idx] + (1.0f - gv) * xs[idx];
}

// -------------------------------------------------------------- launcher
static void launch_gemm(hipStream_t s, const float* A, const float* W,
                        const float* bias, float* C, int M, int N, int K,
                        int lda, int ldw, int ldc,
                        long sA, long sW, long sC, int batch,
                        int act, int out_nchw) {
  int ntn = (N + 31) >> 5;
  int tiles = (M >> 4) * ntn;
  dim3 grid((tiles + 7) / 8, batch);
  hipLaunchKernelGGL(srcm_wmma_gemm, grid, dim3(256), 0, s,
                     A, W, bias, C, M, N, K, lda, ldw, ldc, sA, sW, sC,
                     act, out_nchw);
}

extern "C" void kernel_launch(void* const* d_in, const int* in_sizes, int n_in,
                              void* d_out, int out_size, void* d_ws, size_t ws_size,
                              hipStream_t stream) {
  const float* x        = (const float*)d_in[0];
  const float* pos      = (const float*)d_in[1];
  const float* norm_w   = (const float*)d_in[2];
  const float* norm_b   = (const float*)d_in[3];
  const float* gate_w   = (const float*)d_in[4];
  const float* gate_b   = (const float*)d_in[5];
  const float* proj_w   = (const float*)d_in[6];
  const float* proj_b   = (const float*)d_in[7];
  const float* in_w     = (const float*)d_in[8];
  const float* conv_w_f = (const float*)d_in[9];
  const float* conv_b_f = (const float*)d_in[10];
  const float* xpw_f    = (const float*)d_in[11];
  const float* dtw_f    = (const float*)d_in[12];
  const float* dtb_f    = (const float*)d_in[13];
  const float* Alog_f   = (const float*)d_in[14];
  const float* D_f      = (const float*)d_in[15];
  const float* conv_w_b = (const float*)d_in[16];
  const float* conv_b_b = (const float*)d_in[17];
  const float* xpw_b    = (const float*)d_in[18];
  const float* dtw_b    = (const float*)d_in[19];
  const float* dtb_b    = (const float*)d_in[20];
  const float* Alog_b   = (const float*)d_in[21];
  const float* D_b      = (const float*)d_in[22];
  const float* out_w    = (const float*)d_in[23];
  float* out = (float*)d_out;

  // workspace layout (f32 elements)
  float* ws = (float*)d_ws;
  const size_t SZ_TOKC = (size_t)NTOK * CDIM;                 // 1,048,576
  float* xs   = ws;                      // xs
  float* xn   = xs  + SZ_TOKC;           // layernorm output
  float* gate = xn  + SZ_TOKC;           // sigmoid gate
  float* xz   = gate + SZ_TOKC;          // [g][4096][256]
  float* xc   = xz  + (size_t)NGROUP * SZ_TOKC;               // [gd][4096][128]
  float* p    = xc  + (size_t)NGROUP * 2 * NTOK * D_INNER;    // [gd][4096][36]
  float* dtb_ = p   + (size_t)NGROUP * 2 * NTOK * 36;         // dt
  float* yb   = dtb_ + (size_t)NGROUP * 2 * NTOK * D_INNER;   // [g][4096][128]
  float* xm   = yb  + (size_t)NGROUP * NTOK * D_INNER;        // x_mamba
  float* xo   = xm  + SZ_TOKC;                                // x_out

  // 1. prep + layernorm
  hipLaunchKernelGGL(srcm_prep_ln, dim3(NTOK), dim3(256), 0, stream,
                     x, pos, norm_w, norm_b, xs, xn);

  // 2. gate = sigmoid(xn @ gate_w^T + gate_b)
  launch_gemm(stream, xn, gate_w, gate_b, gate, NTOK, CDIM, CDIM,
              CDIM, CDIM, CDIM, 0, 0, 0, 1, /*act*/1, 0);

  // 3. in_proj per group: xn[:, g*64:(g+1)*64] @ in_w[g]^T -> xz[g]
  launch_gemm(stream, xn, in_w, nullptr, xz, NTOK, 2 * D_INNER, D_MODEL,
              CDIM, D_MODEL, CDIM,
              /*sA*/D_MODEL, /*sW*/(long)2 * D_INNER * D_MODEL,
              /*sC*/(long)NTOK * CDIM, NGROUP, 0, 0);

  // 4. causal conv + SiLU (both directions)
  hipLaunchKernelGGL(srcm_conv_silu, dim3((NGROUP * 2 * NTOK * D_INNER) / 256),
                     dim3(256), 0, stream, xz, conv_w_f, conv_b_f,
                     conv_w_b, conv_b_b, xc);

  // 5. x_proj: xc[gd] @ xpw^T  (N=36, padded to WMMA tiles)
  launch_gemm(stream, xc, xpw_f, nullptr, p, NTOK, 36, D_INNER,
              D_INNER, D_INNER, 36,
              (long)2 * NTOK * D_INNER, (long)36 * D_INNER,
              (long)2 * NTOK * 36, NGROUP, 0, 0);               // dir f (gd even)
  launch_gemm(stream, xc + (size_t)NTOK * D_INNER, xpw_b, nullptr,
              p + (size_t)NTOK * 36, NTOK, 36, D_INNER,
              D_INNER, D_INNER, 36,
              (long)2 * NTOK * D_INNER, (long)36 * D_INNER,
              (long)2 * NTOK * 36, NGROUP, 0, 0);               // dir b (gd odd)

  // 6. dt = softplus(p[:, :4] @ dtw^T + dtb)
  hipLaunchKernelGGL(srcm_dt, dim3((NGROUP * 2 * NTOK * D_INNER) / 256),
                     dim3(256), 0, stream, p, dtw_f, dtb_f, dtw_b, dtb_b, dtb_);

  // 7. selective scans: forward writes, backward accumulates
  hipLaunchKernelGGL(srcm_scan, dim3(8), dim3(256), 0, stream,
                     xc, p, dtb_, xz, Alog_f, D_f, yb, 0);
  hipLaunchKernelGGL(srcm_scan, dim3(8), dim3(256), 0, stream,
                     xc, p, dtb_, xz, Alog_b, D_b, yb, 1);

  // 8. out_proj per group: y[g] @ out_w[g]^T -> x_mamba[:, g*64:]
  launch_gemm(stream, yb, out_w, nullptr, xm, NTOK, D_MODEL, D_INNER,
              D_INNER, D_INNER, CDIM,
              (long)NTOK * D_INNER, (long)D_MODEL * D_INNER,
              /*sC*/D_MODEL, NGROUP, 0, 0);

  // 9. combine: gate * x_mamba + (1-gate) * xs
  hipLaunchKernelGGL(srcm_combine, dim3(NTOK), dim3(256), 0, stream,
                     gate, xm, xs, xo);

  // 10. final proj with NCHW-transposed store straight into d_out
  launch_gemm(stream, xo, proj_w, proj_b, out, NTOK, CDIM, CDIM,
              CDIM, CDIM, CDIM, 0, 0, 0, 1, 0, /*nchw*/1);
}
